// LinearLayerTT_7739531067549
// MI455X (gfx1250) — matrix operand order repaired and checked by reading, not compiled, not run on hardware
//
#include <hip/hip_runtime.h>

// ---------------------------------------------------------------------------
// TT-dense matvec, fused per-batch-element, bf16 WMMA with f32 accumulation.
// One workgroup (8 wave32) per batch element; all intermediates in LDS.
// Fragment loads are 2x ds_load_b128 each; invariant fragments hoisted;
// inner tile loops unrolled x2 for two concurrent accumulator chains.
// ---------------------------------------------------------------------------

typedef __attribute__((ext_vector_type(16))) __bf16       v16bf;
typedef __attribute__((ext_vector_type(8)))  float        v8f;
typedef __attribute__((ext_vector_type(8)))  unsigned int v8u;
typedef __attribute__((ext_vector_type(4)))  unsigned int v4u;
typedef __attribute__((ext_vector_type(4)))  float        v4f;

// A-matrix (16x32 bf16) per CDNA5 ISA 7.12.2: lane holds K = lh*8 + {0..7}
// in elems 0..7 and K = lh*8 + 16 + {0..7} in elems 8..15 (lh = lane>>4).
// Two contiguous 16B runs when K is innermost in LDS.
static __device__ __forceinline__ v16bf frag_a(const __bf16* row, int lh) {
  const __bf16* p = row + lh * 8;
  v4u lo = *(const v4u*)(p);
  v4u hi = *(const v4u*)(p + 16);
  return __builtin_bit_cast(
      v16bf, __builtin_shufflevector(lo, hi, 0, 1, 2, 3, 4, 5, 6, 7));
}
// B-matrix (32x16 bf16): lane holds K = lh*8 + {0..7} in elems 0..7 and
// K = lh*8 + 16 + {0..7} in elems 8..15 (same order as A).
static __device__ __forceinline__ v16bf frag_b(const __bf16* col, int lh) {
  const __bf16* p = col + lh * 8;
  v4u lo = *(const v4u*)(p);
  v4u hi = *(const v4u*)(p + 16);
  return __builtin_bit_cast(
      v16bf, __builtin_shufflevector(lo, hi, 0, 1, 2, 3, 4, 5, 6, 7));
}
static __device__ __forceinline__ int kb_of(int lane, int e) {  // stage-3 B3
  return ((lane >> 4) << 3) + (e < 8 ? e : e + 8);
}

#define WMMA_BF16(a, b, c)                                                   \
  __builtin_amdgcn_wmma_f32_16x16x32_bf16(false, (a), false, (b), (short)0,  \
                                          (c), false, false)

__global__ __launch_bounds__(256) void tt_wmma_kernel(
    const float* __restrict__ x,  const float* __restrict__ c0,
    const float* __restrict__ c1, const float* __restrict__ c2,
    const float* __restrict__ c3, const float* __restrict__ bias,
    float* __restrict__ out) {
  extern __shared__ __bf16 lds[];
  // LDS map (bf16 elements), total 149504 elems = 292 KB:
  __bf16* Y0  = lds;            // 65536: [n1=8][m3m4=64][p=(r1*8+m2)=128]
  __bf16* Y2  = lds;            // 65536: [row=(n1n2n3)=512][p=(r3*8+m4)=128]
  __bf16* Y1  = lds + 65536;    // 65536: [g=(n1n2)=64][m4=8][p=(r2*8+m3)=128]
  __bf16* XBp = lds + 65536;    // 16384: [m2m3m4=512][K=32 pad]  (inside Y1)
  __bf16* A0p = lds + 81920;    //  4096: [q=(n1,r1)=128][K=32 pad] (inside Y1)
  __bf16* B3p = lds + 131072;   //  2048: [p=(r3*8+m4)=128][n4=16 pad]
  __bf16* AC  = lds + 133120;   // 16384: [q=128][p=128]  core1, then core2

  const int tid  = threadIdx.x;
  const int wave = tid >> 5;
  const int lane = tid & 31;
  const int lr   = lane & 15;   // N / row-within-tile lane index
  const int lh   = lane >> 4;   // lane half
  const int b    = blockIdx.x;
  const __bf16 zero = (__bf16)0.0f;

  // ---- Prologue pass 1: zero the padded operand regions -------------------
  for (int i = tid; i < 16384; i += 256) XBp[i] = zero;
  for (int i = tid; i < 4096; i += 256) A0p[i] = zero;
  for (int i = tid; i < 2048; i += 256) B3p[i] = zero;
  __syncthreads();

  // ---- Prologue pass 2: coalesced global reads, bf16 scatter into LDS -----
  const v4f* xb4 = (const v4f*)(x + (size_t)b * 4096);
  for (int i4 = tid; i4 < 1024; i4 += 256) {  // x[m1][col] -> XBp[col][m1]
    v4f v = xb4[i4];
    int i = i4 * 4, m1 = i >> 9, col = i & 511;
#pragma unroll
    for (int j = 0; j < 4; ++j) XBp[(col + j) * 32 + m1] = (__bf16)v[j];
  }
  for (int s = tid; s < 1024; s += 256) {
    {  // c0[s]: s = (n1*8+m1)*16+r1 -> A0p[(n1*16+r1)*32 + m1]
      int r1 = s & 15, m1 = (s >> 4) & 7, n1 = s >> 7;
      A0p[(n1 * 16 + r1) * 32 + m1] = (__bf16)c0[s];
    }
    {  // c3[s]: s = (r3*8+n4)*8+m4 -> B3p[(r3*8+m4)*16 + n4]
      int m4 = s & 7, n4 = (s >> 3) & 7, r3 = s >> 6;
      B3p[(r3 * 8 + m4) * 16 + n4] = (__bf16)c3[s];
    }
  }
  for (int s = tid; s < 16384; s += 256) {
    // c1[s]: s = ((r1*8+n2)*8+m2)*16+r2 -> AC[(n2*16+r2)*128 + r1*8+m2]
    int r2 = s & 15, m2 = (s >> 4) & 7, n2 = (s >> 7) & 7, r1 = s >> 10;
    AC[(n2 * 16 + r2) * 128 + r1 * 8 + m2] = (__bf16)c1[s];
  }
  __syncthreads();

  // ---- Stage 0: Y0[q=(n1,r1)=128, r=(m2,m3,m4)=512] = W0^T(128x8)*X(8x512)
  {
    const v16bf a0 = frag_a(&A0p[(wave * 16 + lr) * 32], lh);
#pragma unroll 2
    for (int tn = 0; tn < 32; ++tn) {
      v8f c = {0.f, 0.f, 0.f, 0.f, 0.f, 0.f, 0.f, 0.f};
      c = WMMA_BF16(a0, frag_b(&XBp[(tn * 16 + lr) * 32], lh), c);
#pragma unroll
      for (int v = 0; v < 8; ++v) {  // D(M,N) -> Y0[n1][m3m4][r1*8+m2]
        int M = wave * 16 + v + 8 * lh, N = tn * 16 + lr;
        int n1 = M >> 4, r1 = M & 15, m2 = N >> 6, m34 = N & 63;
        Y0[(n1 * 64 + m34) * 128 + r1 * 8 + m2] = (__bf16)c[v];
      }
    }
  }
  __syncthreads();

  // ---- Stage 1: per n1: Y1[q=(n2,r2)=128, (m3,m4)=64] = W1^T * Y0', K=128
  {
    v16bf a1[4];
#pragma unroll
    for (int ks = 0; ks < 4; ++ks)
      a1[ks] = frag_a(&AC[(wave * 16 + lr) * 128 + ks * 32], lh);
#pragma unroll 2
    for (int u = 0; u < 32; ++u) {  // 8 n1 x 4 N-tiles
      int n1 = u >> 2, tn = u & 3;
      const __bf16* bcol = &Y0[(n1 * 64 + tn * 16 + lr) * 128];
      v8f c = {0.f, 0.f, 0.f, 0.f, 0.f, 0.f, 0.f, 0.f};
#pragma unroll
      for (int ks = 0; ks < 4; ++ks)
        c = WMMA_BF16(a1[ks], frag_b(bcol + ks * 32, lh), c);
#pragma unroll
      for (int v = 0; v < 8; ++v) {  // -> Y1[(n1,n2)][m4][r2*8+m3]
        int M = wave * 16 + v + 8 * lh, N = tn * 16 + lr;
        int n2 = M >> 4, r2 = M & 15, m3 = N >> 3, m4 = N & 7;
        Y1[((n1 * 8 + n2) * 8 + m4) * 128 + r2 * 8 + m3] = (__bf16)c[v];
      }
    }
  }
  __syncthreads();

  // swap operand slot: AC = W2^T: [q=(n3,r3)][p=(r2,m3)]
  for (int s = tid; s < 16384; s += 256) {
    // c2[s]: s = ((r2*8+n3)*8+m3)*16+r3 -> AC[(n3*16+r3)*128 + r2*8+m3]
    int r3 = s & 15, m3 = (s >> 4) & 7, n3 = (s >> 7) & 7, r2 = s >> 10;
    AC[(n3 * 16 + r3) * 128 + r2 * 8 + m3] = (__bf16)c2[s];
  }
  __syncthreads();

  // ---- Stage 2: per group g=(n1,n2): [128 x 8], K=128; two groups per
  // N=16 tile (columns independent). Writes Y2 (reuses Y0 region).
  {
    v16bf a2[4];
#pragma unroll
    for (int ks = 0; ks < 4; ++ks)
      a2[ks] = frag_a(&AC[(wave * 16 + lr) * 128 + ks * 32], lh);
    const int m4 = lr & 7;
#pragma unroll 2
    for (int j = 0; j < 32; ++j) {  // group pairs
      int g = 2 * j + (lr >> 3);    // this lane's group
      const __bf16* bcol = &Y1[(g * 8 + m4) * 128];
      v8f c = {0.f, 0.f, 0.f, 0.f, 0.f, 0.f, 0.f, 0.f};
#pragma unroll
      for (int ks = 0; ks < 4; ++ks)
        c = WMMA_BF16(a2[ks], frag_b(bcol + ks * 32, lh), c);
#pragma unroll
      for (int v = 0; v < 8; ++v) {  // -> Y2[(g*8+n3)][r3*8+m4]
        int M = wave * 16 + v + 8 * lh;
        int n3 = M >> 4, r3 = M & 15;
        Y2[(g * 8 + n3) * 128 + r3 * 8 + m4] = (__bf16)c[v];
      }
    }
  }
  __syncthreads();

  // ---- Stage 3: Y[512 rows=(n1,n2,n3), n4=8] = Y2(512x128) * W3(128x8)
  // N padded to 16 in LDS; B fragments hoisted. Fused bias + store.
  {
    v16bf b3[4];
#pragma unroll
    for (int ks = 0; ks < 4; ++ks) {
      v16bf bm;
#pragma unroll
      for (int e = 0; e < 16; ++e)
        bm[e] = B3p[(ks * 32 + kb_of(lane, e)) * 16 + lr];
      b3[ks] = bm;
    }
    float* ob = out + (size_t)b * 4096;
#pragma unroll 2
    for (int tm = wave; tm < 32; tm += 8) {
      const __bf16* arow = &Y2[(tm * 16 + lr) * 128];
      v8f c = {0.f, 0.f, 0.f, 0.f, 0.f, 0.f, 0.f, 0.f};
#pragma unroll
      for (int ks = 0; ks < 4; ++ks)
        c = WMMA_BF16(frag_a(arow + ks * 32, lh), b3[ks], c);
      if (lr < 8) {
#pragma unroll
        for (int v = 0; v < 8; ++v) {
          int R = tm * 16 + v + 8 * lh;  // (n1,n2,n3)
          int o = R * 8 + lr;            // + n4
          ob[o] = c[v] + bias[o];
        }
      }
    }
  }
}

// ---------------------------------------------------------------------------

extern "C" void kernel_launch(void* const* d_in, const int* in_sizes, int n_in,
                              void* d_out, int out_size, void* d_ws,
                              size_t ws_size, hipStream_t stream) {
  (void)n_in; (void)out_size; (void)d_ws; (void)ws_size;
  const float* x  = (const float*)d_in[0];
  const float* c0 = (const float*)d_in[1];
  const float* c1 = (const float*)d_in[2];
  const float* c2 = (const float*)d_in[3];
  const float* c3 = (const float*)d_in[4];
  const float* bs = (const float*)d_in[5];
  float* out = (float*)d_out;

  const int batch = in_sizes[0] / 4096;              // 8192
  const size_t lds_bytes = 149504 * sizeof(__bf16);  // 292 KB of 320 KB WGP LDS
  (void)hipFuncSetAttribute((const void*)tt_wmma_kernel,
                            hipFuncAttributeMaxDynamicSharedMemorySize,
                            (int)lds_bytes);
  hipLaunchKernelGGL(tt_wmma_kernel, dim3(batch), dim3(256), lds_bytes, stream,
                     x, c0, c1, c2, c3, bs, out);
}
